// SimpleGCN_8787503087823
// MI455X (gfx1250) — matrix operand order repaired and checked by reading, not compiled, not run on hardware
//
#include <hip/hip_runtime.h>
#include <hip/hip_bf16.h>

// ---------------------------------------------------------------------------
// GCN pipeline for MI455X (gfx1250, wave32, WMMA).
//   h  = spmm(A, x)            [N,512]   edge-parallel atomic scatter
//   h1 = h @ W1 + b1           [N,256]   bf16 WMMA, f32 accum
//   s2 = spmm(A, h1)           [N,256]
//   t1 = s2 @ W2 + b2          [N,64]    bf16 WMMA
//   t2 = h1 @ Wres + bres      [N,64]    bf16 WMMA
//   out = relu(t1) + t2        [N,64]
// ---------------------------------------------------------------------------

typedef __attribute__((ext_vector_type(16))) __bf16 v16bf;
typedef __attribute__((ext_vector_type(4)))  __bf16 v4bf;
typedef __attribute__((ext_vector_type(4)))  float  v4f;
typedef __attribute__((ext_vector_type(8)))  float  v8f;

// -------------------------------- zero fill --------------------------------
__global__ __launch_bounds__(256) void zero_kernel(float* __restrict__ p, long long n) {
  long long i = ((long long)blockIdx.x * blockDim.x + threadIdx.x) * 4;
  if (i + 3 < n) {
    *(float4*)(p + i) = make_float4(0.f, 0.f, 0.f, 0.f);
  } else {
    for (long long j = i; j < n; ++j) p[j] = 0.f;
  }
}

// --------------------------------- SpMM ------------------------------------
// out[rows[e], :] += vals[e] * x[cols[e], :]
// One float4 feature-chunk per thread; the (F/4) threads of an edge are
// consecutive -> the 2KB row gather is fully coalesced. Scatter via
// global_atomic_add_f32 (no-return, STOREcnt).
__global__ __launch_bounds__(256) void spmm_kernel(
    const int* __restrict__ rows, const int* __restrict__ cols,
    const float* __restrict__ vals, const float* __restrict__ x,
    float* __restrict__ out, int nEdges, int F, int chunkLog) {
  long long tid = (long long)blockIdx.x * blockDim.x + threadIdx.x;
  int e = (int)(tid >> chunkLog);
  if (e >= nEdges) return;
  int c = (int)(tid & ((1 << chunkLog) - 1));
  int r  = rows[e];
  int cc = cols[e];
  float v = vals[e];
  const float4 xv = *(const float4*)(x + (size_t)cc * F + (c << 2));
  float* dst = out + (size_t)r * F + (c << 2);
  atomicAdd(dst + 0, v * xv.x);
  atomicAdd(dst + 1, v * xv.y);
  atomicAdd(dst + 2, v * xv.z);
  atomicAdd(dst + 3, v * xv.w);
}

// ---------------------------- WMMA GEMM + bias -----------------------------
// C[M,F] = A[M,K] @ W[K,F] + bias.  fp32 in/out, bf16 WMMA math, f32 accum.
// Block: 256 threads = 8 waves -> 64(M) x 64(N) tile, K stepped by 32.
// Each wave: one 16-row subtile x two 16-col subtiles (A fragment reused,
// 2 v_wmma per K-step). Double-buffered LDS pipeline with one barrier/iter;
// f32->bf16 via native packed converts (__builtin_convertvector).
#define BM 64
#define BN 64
#define BK 32

__global__ __launch_bounds__(256) void gemm_bias_wmma(
    const float* __restrict__ A, const float* __restrict__ W,
    const float* __restrict__ bias, float* __restrict__ C,
    int M, int K, int F) {
  __shared__ __bf16 sA[2][BM][BK + 4];  // 72B row stride: 8B-aligned v4bf stores
  __shared__ __bf16 sB[2][BK][BN + 4];  // 136B row stride

  const int t    = threadIdx.x;
  const int lane = t & 31;
  const int wave = t >> 5;
  const int wm   = wave >> 1;   // 0..3 : M sub-tile
  const int wn   = wave & 1;    // 0..1 : pair of N sub-tiles (32 cols)
  const int hi   = lane >> 4;   // half-wave select
  const int lo   = lane & 15;

  const int rowBase = blockIdx.y * BM;
  const int colBase = blockIdx.x * BN;

  v4f ra[2], rb[2];  // in-flight staging registers

  auto loadTiles = [&](int k0) {
#pragma unroll
    for (int j = 0; j < 2; ++j) {           // A tile 64x32, 2 float4/thread
      int i  = t + j * 256;
      int mi = i >> 3, kc = (i & 7) << 2;
      int gm = rowBase + mi;
      int gms = (gm < M) ? gm : (M - 1);    // clamp: branch-free, always valid
      ra[j] = *(const v4f*)(A + (size_t)gms * K + k0 + kc);
    }
#pragma unroll
    for (int j = 0; j < 2; ++j) {           // B tile 32x64, 2 float4/thread
      int i  = t + j * 256;
      int ki = i >> 4, nc = (i & 15) << 2;
      rb[j] = *(const v4f*)(W + (size_t)(k0 + ki) * F + colBase + nc);
    }
  };
  auto storeTiles = [&](int p) {
#pragma unroll
    for (int j = 0; j < 2; ++j) {
      int i  = t + j * 256;
      int mi = i >> 3, kc = (i & 7) << 2;
      float msk = ((rowBase + mi) < M) ? 1.f : 0.f;  // zero OOB rows
      v4f v = ra[j] * msk;
      *(v4bf*)&sA[p][mi][kc] = __builtin_convertvector(v, v4bf);
    }
#pragma unroll
    for (int j = 0; j < 2; ++j) {
      int i  = t + j * 256;
      int ki = i >> 4, nc = (i & 15) << 2;
      *(v4bf*)&sB[p][ki][nc] = __builtin_convertvector(rb[j], v4bf);
    }
  };

  v8f acc0 = {}, acc1 = {};

  loadTiles(0);
  storeTiles(0);
  __syncthreads();

  int p = 0;
  for (int k0 = 0; k0 < K; k0 += BK) {
    const bool more = (k0 + BK) < K;        // uniform
    if (more) loadTiles(k0 + BK);           // issue next tile's loads early

    // Fragments per CDNA5 ISA layouts (16-bit A 16x32, B 32x16).
    union { unsigned int u[8]; v16bf v; } af, bf0, bf1;
    const int am = wm * 16 + lo;
#pragma unroll
    for (int j = 0; j < 4; ++j)   // VGPR 0..3: K = hi*8 + {0,2,4,6}(+1)
      af.u[j] = *(const unsigned int*)&sA[p][am][hi * 8 + 2 * j];
#pragma unroll
    for (int j = 0; j < 4; ++j)   // VGPR 4..7: K = 16 + hi*8 + {0,2,4,6}(+1)
      af.u[4 + j] = *(const unsigned int*)&sA[p][am][16 + hi * 8 + 2 * j];
    const int bn = wn * 32;
#pragma unroll
    for (int s = 0; s < 16; ++s) {  // N = lo; K = hi*8 + s (s<8), 16 + hi*8 + (s-8) (s>=8)
      const int kk = (s < 8) ? (hi * 8 + s) : (16 + hi * 8 + (s - 8));
      bf0.v[s] = sB[p][kk][bn + lo];
      bf1.v[s] = sB[p][kk][bn + 16 + lo];
    }

    acc0 = __builtin_amdgcn_wmma_f32_16x16x32_bf16(
        false, af.v, false, bf0.v, (short)0, acc0, false, false);
    acc1 = __builtin_amdgcn_wmma_f32_16x16x32_bf16(
        false, af.v, false, bf1.v, (short)0, acc1, false, false);

    if (more) storeTiles(p ^ 1);            // fill other buffer
    __syncthreads();                        // one barrier per K-step
    p ^= 1;
  }

  // C layout: VGPR r -> M = r + 8*hi ; N = lo
  const int gn0 = colBase + wn * 32 + lo;
  const float bv0 = bias[gn0];
  const float bv1 = bias[gn0 + 16];
  size_t cbase = (size_t)(rowBase + wm * 16 + 8 * hi) * F + gn0;
  if (rowBase + BM <= M) {                  // uniform: interior block, no guards
#pragma unroll
    for (int r = 0; r < 8; ++r) {
      C[cbase]      = acc0[r] + bv0;
      C[cbase + 16] = acc1[r] + bv1;
      cbase += F;
    }
  } else {                                  // tail block only
#pragma unroll
    for (int r = 0; r < 8; ++r) {
      int gm = rowBase + wm * 16 + r + 8 * hi;
      if (gm < M) {
        C[(size_t)gm * F + gn0]      = acc0[r] + bv0;
        C[(size_t)gm * F + gn0 + 16] = acc1[r] + bv1;
      }
    }
  }
}

// ------------------------------ relu + add ---------------------------------
__global__ __launch_bounds__(256) void relu_add_kernel(
    const float* __restrict__ a, const float* __restrict__ b,
    float* __restrict__ o, long long n) {
  long long i = (long long)blockIdx.x * blockDim.x + threadIdx.x;
  if (i < n) o[i] = fmaxf(a[i], 0.f) + b[i];
}

// ---------------------------------------------------------------------------
extern "C" void kernel_launch(void* const* d_in, const int* in_sizes, int n_in,
                              void* d_out, int out_size, void* d_ws, size_t ws_size,
                              hipStream_t stream) {
  const float* x    = (const float*)d_in[0];
  const int*   rows = (const int*)d_in[1];
  const int*   cols = (const int*)d_in[2];
  const float* vals = (const float*)d_in[3];
  const float* W1   = (const float*)d_in[4];
  const float* b1   = (const float*)d_in[5];
  const float* W2   = (const float*)d_in[6];
  const float* b2   = (const float*)d_in[7];
  const float* Wres = (const float*)d_in[8];
  const float* bres = (const float*)d_in[9];
  float* out = (float*)d_out;

  const int E     = in_sizes[1];
  const int NHID  = in_sizes[5];          // 256
  const int NCLS  = in_sizes[7];          // 64
  const int NFEAT = in_sizes[4] / NHID;   // 512
  const int N     = in_sizes[0] / NFEAT;  // 100000

  // workspace layout: accumulators (hA, s2) contiguous for one zero pass
  float* hA = (float*)d_ws;                 // [N, NFEAT] spmm1 accum
  float* s2 = hA + (size_t)N * NFEAT;       // [N, NHID]  spmm2 accum
  float* h1 = s2 + (size_t)N * NHID;        // [N, NHID]  hidden
  float* t1 = h1 + (size_t)N * NHID;        // [N, NCLS]  s2@W2+b2
  float* t2 = t1 + (size_t)N * NCLS;        // [N, NCLS]  h1@Wres+bres

  // 1) zero atomic accumulators
  {
    long long nz = (long long)N * (NFEAT + NHID);
    long long threads = (nz + 3) / 4;
    long long blocks  = (threads + 255) / 256;
    zero_kernel<<<(unsigned)blocks, 256, 0, stream>>>(hA, nz);
  }

  // 2) spmm1: hA = A @ x   (F = 512, 128 float4-chunks/edge)
  {
    int cl = 31 - __builtin_clz((unsigned)(NFEAT >> 2));
    long long threads = (long long)E << cl;
    long long blocks  = (threads + 255) / 256;
    spmm_kernel<<<(unsigned)blocks, 256, 0, stream>>>(rows, cols, vals, x, hA, E, NFEAT, cl);
  }

  // 3) h1 = hA @ W1 + b1
  {
    dim3 g(NHID / BN, (N + BM - 1) / BM);
    gemm_bias_wmma<<<g, 256, 0, stream>>>(hA, W1, b1, h1, N, NFEAT, NHID);
  }

  // 4) spmm2: s2 = A @ h1   (F = 256)
  {
    int cl = 31 - __builtin_clz((unsigned)(NHID >> 2));
    long long threads = (long long)E << cl;
    long long blocks  = (threads + 255) / 256;
    spmm_kernel<<<(unsigned)blocks, 256, 0, stream>>>(rows, cols, vals, h1, s2, E, NHID, cl);
  }

  // 5) t1 = s2 @ W2 + b2 ;  t2 = h1 @ Wres + bres
  {
    dim3 g(NCLS / BN, (N + BM - 1) / BM);
    gemm_bias_wmma<<<g, 256, 0, stream>>>(s2, W2, b2, t1, N, NHID, NCLS);
    gemm_bias_wmma<<<g, 256, 0, stream>>>(h1, Wres, bres, t2, N, NHID, NCLS);
  }

  // 6) out = relu(t1) + t2
  {
    long long n = (long long)N * NCLS;
    long long blocks = (n + 255) / 256;
    relu_add_kernel<<<(unsigned)blocks, 256, 0, stream>>>(t1, t2, out, n);
  }
}
